// Actor_58394375356775
// MI455X (gfx1250) — compile-verified, matrix operand order checked by reading, not run on hardware
//
#include <hip/hip_runtime.h>
#include <hip/hip_bf16.h>
#include <math.h>

// Problem dims (match reference)
#define B_  64
#define T_  64
#define V_  32000
#define H_  512
#define E_  256
#define G3H (3 * H_)   // 1536

typedef __attribute__((ext_vector_type(16))) __bf16 v16bf;
typedef __attribute__((ext_vector_type(8)))  float  v8f;
typedef __attribute__((ext_vector_type(8)))  int    v8i;

// ---------------- helpers ----------------

__device__ __forceinline__ unsigned short f2bf(float f) {
    unsigned int x = __float_as_uint(f);
    // round-to-nearest-even bf16 truncation
    unsigned int r = (x + 0x7FFFu + ((x >> 16) & 1u)) >> 16;
    return (unsigned short)r;
}

__device__ __forceinline__ unsigned int hash3(unsigned int a, unsigned int b, unsigned int c) {
    unsigned int h = a * 0x9E3779B1u + b * 0x85EBCA77u + c * 0xC2B2AE3Du + 0x27220A95u;
    h ^= h >> 15; h *= 0x2C1B3C6Du;
    h ^= h >> 12; h *= 0x297A2D39u;
    h ^= h >> 15;
    return h;
}

__device__ __forceinline__ float u01(unsigned int h) {
    return (float)(h >> 8) * (1.0f / 16777216.0f);
}

// Load one 32-deep K-chunk of fragments (A from LDS, two B tiles from global).
__device__ __forceinline__ void load_frags(const unsigned int* __restrict__ sArow,
                                           const unsigned short* __restrict__ Brow0,
                                           const unsigned short* __restrict__ Brow1,
                                           const int* kp, int kbase,
                                           v8i& a, v8i& b0, v8i& b1) {
#pragma unroll
    for (int j = 0; j < 8; ++j) {
        const int k = kbase + kp[j];
        a[j]  = sArow[k >> 1];
        b0[j] = *(const unsigned int*)(Brow0 + k);
        b1[j] = *(const unsigned int*)(Brow1 + k);
    }
}

// ---------------- generic bf16 WMMA GEMM ----------------
// C[64 x N] = A[64 x K] * Bw[N x K]^T + bias[N]   (bf16 in, f32 accumulate/out)
// Block = 128 threads = 4 waves; wave w owns m-tile w (rows 16w..16w+15);
// blockIdx.x owns a 32-column block (two 16-wide n-tiles per wave sharing the
// A fragment). A is staged in LDS once per block. The K loop is unrolled 2x
// with ping-pong fragment sets (static register roles -> no rotate copies):
//   load B-set(k+32) ; WMMA A-set x2 ; load A-set(k+64) ; WMMA B-set x2
// Requires K % 64 == 0 (holds for K = 256 and K = 512).
__global__ void __launch_bounds__(128)
wmma_mm_bf16(const unsigned short* __restrict__ A,   // 64 x K
             const unsigned short* __restrict__ Bw,  // N x K  (weight[n][k])
             const float* __restrict__ bias,         // N
             float* __restrict__ C,                  // 64 x N
             int K, int N)
{
    extern __shared__ unsigned int sA32[];           // 64*K/2 dwords
    const int tid = threadIdx.x;

    // cooperative stage of A into LDS (dword granularity; K is even)
    const unsigned int* A32 = (const unsigned int*)A;
    const int ndw = (64 * K) >> 1;
    for (int i = tid; i < ndw; i += 128) sA32[i] = A32[i];
    __syncthreads();

    const int wave = tid >> 5;          // 0..3 -> m tile
    const int lane = tid & 31;
    const int g    = lane >> 4;         // lane half (ISA 16-bit layout)
    const int ln16 = lane & 15;
    const int m0   = wave << 4;
    const int n0   = blockIdx.x << 5;   // 32 columns per block

    const unsigned short* Brow0 = Bw + (size_t)(n0 + ln16) * (size_t)K;
    const unsigned short* Brow1 = Brow0 + (size_t)16 * (size_t)K;
    const unsigned int*   sArow = sA32 + (((m0 + ln16) * K) >> 1);

    // K-pair offsets for this lane (ISA 16-bit A/B fragment layout:
    //   VGPR j<4 : K = g*8 + 2j ; VGPR j>=4 : K = 16 + g*8 + 2(j-4))
    int kp[8];
#pragma unroll
    for (int j = 0; j < 8; ++j)
        kp[j] = (j < 4) ? (g * 8 + 2 * j) : (16 + g * 8 + 2 * (j - 4));

    v8f acc0 = {}, acc1 = {};
    v8i aA, b0A, b1A, aB, b0B, b1B;

    load_frags(sArow, Brow0, Brow1, kp, 0, aA, b0A, b1A);

    for (int k0 = 0; k0 < K; k0 += 64) {
        __builtin_prefetch(Brow0 + k0 + 128, 0, 1);   // global_prefetch_b8
        __builtin_prefetch(Brow1 + k0 + 128, 0, 1);

        load_frags(sArow, Brow0, Brow1, kp, k0 + 32, aB, b0B, b1B);

        acc0 = __builtin_amdgcn_wmma_f32_16x16x32_bf16(
            false, __builtin_bit_cast(v16bf, aA), false,
            __builtin_bit_cast(v16bf, b0A), (short)0, acc0, false, false);
        acc1 = __builtin_amdgcn_wmma_f32_16x16x32_bf16(
            false, __builtin_bit_cast(v16bf, aA), false,
            __builtin_bit_cast(v16bf, b1A), (short)0, acc1, false, false);

        if (k0 + 64 < K)
            load_frags(sArow, Brow0, Brow1, kp, k0 + 64, aA, b0A, b1A);

        acc0 = __builtin_amdgcn_wmma_f32_16x16x32_bf16(
            false, __builtin_bit_cast(v16bf, aB), false,
            __builtin_bit_cast(v16bf, b0B), (short)0, acc0, false, false);
        acc1 = __builtin_amdgcn_wmma_f32_16x16x32_bf16(
            false, __builtin_bit_cast(v16bf, aB), false,
            __builtin_bit_cast(v16bf, b1B), (short)0, acc1, false, false);
    }

    const int   col0 = n0 + ln16;
    const int   col1 = col0 + 16;
    const float bb0  = bias ? bias[col0] : 0.0f;
    const float bb1  = bias ? bias[col1] : 0.0f;
#pragma unroll
    for (int r = 0; r < 8; ++r) {
        const int row = m0 + r + (g << 3);   // C/D layout: VGPR r, lane-half g
        C[(size_t)row * (size_t)N + col0] = acc0[r] + bb0;
        C[(size_t)row * (size_t)N + col1] = acc1[r] + bb1;
    }
}

// ---------------- one-time conversions / init ----------------

__global__ void f32_to_bf16_kernel(const float* __restrict__ src,
                                   unsigned short* __restrict__ dst, int n) {
    int i = blockIdx.x * blockDim.x + threadIdx.x;
    if (i < n) dst[i] = f2bf(src[i]);
}

__global__ void init_state_kernel(const float* __restrict__ embedding,
                                  float* __restrict__ h,
                                  unsigned short* __restrict__ hbf,
                                  unsigned short* __restrict__ xbf) {
    int i = blockIdx.x * blockDim.x + threadIdx.x;
    if (i < B_ * H_) { h[i] = 0.0f; hbf[i] = f2bf(0.0f); }
    if (i < B_ * E_) {
        int e = i % E_;                       // inputs0[b][e] = embedding[0][e]
        xbf[i] = f2bf(embedding[e]);
    }
}

// ---------------- GRU gates ----------------
__global__ void gru_gates_kernel(const float* __restrict__ gi,   // 64 x 3H
                                 const float* __restrict__ gh,   // 64 x 3H
                                 float* __restrict__ h,          // 64 x H
                                 unsigned short* __restrict__ hbf) {
    int i = blockIdx.x * blockDim.x + threadIdx.x;
    if (i >= B_ * H_) return;
    int b  = i / H_;
    int hh = i % H_;
    const float* gib = gi + (size_t)b * G3H;
    const float* ghb = gh + (size_t)b * G3H;
    float r = 1.0f / (1.0f + expf(-(gib[hh]          + ghb[hh])));
    float z = 1.0f / (1.0f + expf(-(gib[H_ + hh]     + ghb[H_ + hh])));
    float n = tanhf(gib[2 * H_ + hh] + r * ghb[2 * H_ + hh]);
    float hp = h[i];
    float hn = (1.0f - z) * n + z * hp;
    h[i]   = hn;
    hbf[i] = f2bf(hn);
}

// ---------------- softmax row stats (online max + sumexp) ----------------
__global__ void __launch_bounds__(256)
row_stats_kernel(const float* __restrict__ logits,     // 64 x V
                 float* __restrict__ rowmax, float* __restrict__ rowlse) {
    __shared__ float sm[256], ss[256];
    const int b = blockIdx.x, tid = threadIdx.x;
    const float* row = logits + (size_t)b * V_;
    float m = -INFINITY, s = 0.0f;
    for (int v = tid; v < V_; v += 256) {
        float x = row[v];
        if (x > m) { s = s * expf(m - x) + 1.0f; m = x; }
        else         s += expf(x - m);
    }
    sm[tid] = m; ss[tid] = s;
    __syncthreads();
    for (int st = 128; st > 0; st >>= 1) {
        if (tid < st) {
            float m2 = sm[tid + st], s2 = ss[tid + st];
            float M  = fmaxf(sm[tid], m2);
            ss[tid]  = ss[tid] * expf(sm[tid] - M) + s2 * expf(m2 - M);
            sm[tid]  = M;
        }
        __syncthreads();
    }
    if (tid == 0) { rowmax[b] = sm[0]; rowlse[b] = logf(ss[0]); }
}

// ---------------- Gumbel / epsilon-greedy sampling ----------------
__global__ void __launch_bounds__(256)
sample_kernel(const float* __restrict__ logits,
              const float* __restrict__ rowmax, const float* __restrict__ rowlse,
              int t,
              float* __restrict__ out_samples,   // [B][T]
              float* __restrict__ out_corr,      // [B][T]
              float* __restrict__ out_logp,      // [B][T]
              int* __restrict__ sampled) {
    __shared__ float bg[256];
    __shared__ int   bv[256];
    const int b = blockIdx.x, tid = threadIdx.x;
    const float* row = logits + (size_t)b * V_;
    const float mB = rowmax[b], lse = rowlse[b];
    const float log_unif = -logf((float)V_);

    const float u_eps = u01(hash3(0xE15u, (unsigned)t, (unsigned)b));
    const bool  draw  = (0.1f >= u_eps);

    float best  = -INFINITY;
    int   besti = V_;
    for (int v = tid; v < V_; v += 256) {
        float d  = row[v] - mB - lse;
        float de = draw ? log_unif : d;
        float u  = fmaxf(u01(hash3(0x6B1u + (unsigned)t, (unsigned)b, (unsigned)v)), 1e-9f);
        float gs = de - logf(-logf(u));
        if (gs > best || (gs == best && v < besti)) { best = gs; besti = v; }
    }
    bg[tid] = best; bv[tid] = besti;
    __syncthreads();
    for (int st = 128; st > 0; st >>= 1) {
        if (tid < st) {
            if (bg[tid + st] > bg[tid] ||
                (bg[tid + st] == bg[tid] && bv[tid + st] < bv[tid])) {
                bg[tid] = bg[tid + st]; bv[tid] = bv[tid + st];
            }
        }
        __syncthreads();
    }
    if (tid == 0) {
        int vs = bv[0];
        float d    = row[vs] - mB - lse;
        float de   = draw ? log_unif : d;
        float onp  = fminf(fmaxf(expf(d),  1e-8f), 1.0f);
        float offp = fminf(fmaxf(expf(de), 1e-8f), 1.0f);
        out_samples[(size_t)b * T_ + t] = (float)vs;
        out_corr   [(size_t)b * T_ + t] = onp / offp;
        out_logp   [(size_t)b * T_ + t] = d;
        sampled[b] = vs;
    }
}

// ---------------- dists (B,T,V) + prob-mean (T,V) writeout ----------------
__global__ void write_dist_probs_kernel(const float* __restrict__ logits,
                                        const float* __restrict__ rowmax,
                                        const float* __restrict__ rowlse, int t,
                                        float* __restrict__ dists,   // [B][T][V]
                                        float* __restrict__ probs) { // [T][V]
    int v = blockIdx.x * blockDim.x + threadIdx.x;
    if (v >= V_) return;
    float acc = 0.0f;
    for (int b = 0; b < B_; ++b) {
        float d = logits[(size_t)b * V_ + v] - rowmax[b] - rowlse[b];
        dists[((size_t)b * T_ + t) * (size_t)V_ + v] = d;
        acc += expf(d);
    }
    probs[(size_t)t * V_ + v] = acc * (1.0f / (float)B_);
}

// ---------------- next-input embedding gather ----------------
__global__ void gather_embed_kernel(const float* __restrict__ embedding,
                                    const int* __restrict__ sampled,
                                    unsigned short* __restrict__ xbf) {
    int i = blockIdx.x * blockDim.x + threadIdx.x;
    if (i >= B_ * E_) return;
    int b = i / E_, e = i % E_;
    xbf[i] = f2bf(embedding[(size_t)sampled[b] * E_ + e]);
}

// ---------------- host orchestration ----------------
extern "C" void kernel_launch(void* const* d_in, const int* in_sizes, int n_in,
                              void* d_out, int out_size, void* d_ws, size_t ws_size,
                              hipStream_t stream) {
    (void)in_sizes; (void)n_in; (void)out_size; (void)ws_size;

    const float* embedding = (const float*)d_in[0]; // V x E
    const float* w_ih      = (const float*)d_in[1]; // 3H x E
    const float* b_ih      = (const float*)d_in[2]; // 3H
    const float* w_hh      = (const float*)d_in[3]; // 3H x H
    const float* b_hh      = (const float*)d_in[4]; // 3H
    const float* w_out     = (const float*)d_in[5]; // V x H
    const float* b_out     = (const float*)d_in[6]; // V

    // workspace carve-up (~44 MB)
    char* wsp = (char*)d_ws;
    auto carve = [&](size_t bytes) {
        char* p = wsp; wsp += (bytes + 255) & ~(size_t)255; return p;
    };
    unsigned short* w_out_bf = (unsigned short*)carve((size_t)V_ * H_ * 2);
    unsigned short* w_ih_bf  = (unsigned short*)carve((size_t)G3H * E_ * 2);
    unsigned short* w_hh_bf  = (unsigned short*)carve((size_t)G3H * H_ * 2);
    float*          logits   = (float*)carve((size_t)B_ * V_ * 4);
    float*          gi       = (float*)carve((size_t)B_ * G3H * 4);
    float*          gh       = (float*)carve((size_t)B_ * G3H * 4);
    float*          hstate   = (float*)carve((size_t)B_ * H_ * 4);
    unsigned short* h_bf     = (unsigned short*)carve((size_t)B_ * H_ * 2);
    unsigned short* x_bf     = (unsigned short*)carve((size_t)B_ * E_ * 2);
    float*          rowmax   = (float*)carve(B_ * 4);
    float*          rowlse   = (float*)carve(B_ * 4);
    int*            sampled  = (int*)carve(B_ * 4);

    // output carve-up (flat concat of the reference tuple, all f32)
    float* out          = (float*)d_out;
    float* out_samples  = out;                                   // B*T
    float* out_corr     = out + (size_t)B_ * T_;                 // B*T
    float* out_logp     = out + 2 * (size_t)B_ * T_;             // B*T
    float* out_dists    = out + 3 * (size_t)B_ * T_;             // B*T*V
    float* out_probs    = out_dists + (size_t)B_ * T_ * V_;      // T*V

    // one-time: bf16 weight conversions + state init
    {
        int n = V_ * H_;
        f32_to_bf16_kernel<<<(n + 255) / 256, 256, 0, stream>>>(w_out, w_out_bf, n);
        n = G3H * E_;
        f32_to_bf16_kernel<<<(n + 255) / 256, 256, 0, stream>>>(w_ih, w_ih_bf, n);
        n = G3H * H_;
        f32_to_bf16_kernel<<<(n + 255) / 256, 256, 0, stream>>>(w_hh, w_hh_bf, n);
        init_state_kernel<<<(B_ * H_ + 255) / 256, 256, 0, stream>>>(
            embedding, hstate, h_bf, x_bf);
    }

    const size_t ldsE = (size_t)64 * E_ * 2;   // 32 KB  (A stage, K=E)
    const size_t ldsH = (size_t)64 * H_ * 2;   // 64 KB  (A stage, K=H)

    for (int t = 0; t < T_; ++t) {
        // gi = x * w_ih^T + b_ih    (64 x 1536, K=256)
        wmma_mm_bf16<<<G3H / 32, 128, ldsE, stream>>>(x_bf, w_ih_bf, b_ih, gi, E_, G3H);
        // gh = h * w_hh^T + b_hh    (64 x 1536, K=512)
        wmma_mm_bf16<<<G3H / 32, 128, ldsH, stream>>>(h_bf, w_hh_bf, b_hh, gh, H_, G3H);
        // gates -> new hidden (f32 + bf16 copy)
        gru_gates_kernel<<<(B_ * H_ + 255) / 256, 256, 0, stream>>>(gi, gh, hstate, h_bf);
        // logits = h * w_out^T + b_out  (64 x 32000, K=512) -- L2-resident bf16 weights
        wmma_mm_bf16<<<V_ / 32, 128, ldsH, stream>>>(h_bf, w_out_bf, b_out, logits, H_, V_);
        // log-softmax row stats
        row_stats_kernel<<<B_, 256, 0, stream>>>(logits, rowmax, rowlse);
        // sample + per-token outputs
        sample_kernel<<<B_, 256, 0, stream>>>(logits, rowmax, rowlse, t,
                                              out_samples, out_corr, out_logp, sampled);
        // dists + prob-mean writeout (the HBM-heavy 524 MB stream)
        write_dist_probs_kernel<<<(V_ + 255) / 256, 256, 0, stream>>>(
            logits, rowmax, rowlse, t, out_dists, out_probs);
        // next input
        gather_embed_kernel<<<(B_ * E_ + 255) / 256, 256, 0, stream>>>(
            embedding, sampled, x_bf);
    }
}
